// BPEBaselineModel_50938312131193
// MI455X (gfx1250) — compile-verified, matrix operand order checked by reading, not run on hardware
//
#include <hip/hip_runtime.h>
#include <hip/hip_bf16.h>

// ---------------------------------------------------------------------------
// 4-layer transformer forward (D=1024, H=16, HD=64, F=4096, V=32000, B=2,
// T=1024) for MI455X / gfx1250.
//
// v4: attention key-tile loop made block-uniform so K/V tiles are staged
// cooperatively once per block: K via the Tensor Data Mover, V transposed
// into LDS so P*V fragments are contiguous 32B ds_loads (no strided global
// gather). GEMMs keep TDM double-buffered bf16 LDS staging; weights pre-cast
// and transposed to [N,K] bf16 once per call; pure WMMA inner loops.
// ---------------------------------------------------------------------------

typedef __bf16 v16bf __attribute__((ext_vector_type(16)));
typedef __bf16 v8bf  __attribute__((ext_vector_type(8)));
typedef float  v8f   __attribute__((ext_vector_type(8)));
typedef unsigned int u32x4 __attribute__((ext_vector_type(4)));
typedef int          i32x4 __attribute__((ext_vector_type(4)));
typedef int          i32x8 __attribute__((ext_vector_type(8)));

#define LAYERS  4
#define DMODEL  1024
#define NHEAD   16
#define HEADD   64
#define FFN     4096
#define VOCAB   32000
#define BATCH   2
#define SEQ     1024
#define EPSRMS  1e-6f
#define MTOK    (BATCH * SEQ)          // 2048 token rows

static __device__ __forceinline__ v8f wmma_bf16(v16bf a, v16bf b, v8f c) {
    return __builtin_amdgcn_wmma_f32_16x16x32_bf16(
        false, a, false, b, (short)0, c, false, false);
}

// Wave-local LDS ordering fence (softmax scratch is wave-private).
static __device__ __forceinline__ void wave_lds_fence() {
    __builtin_amdgcn_wave_barrier();
    asm volatile("s_wait_dscnt 0x0" ::: "memory");
    __builtin_amdgcn_wave_barrier();
}

static __device__ __forceinline__ void zero8(v8f& v) {
#pragma unroll
    for (int i = 0; i < 8; ++i) v[i] = 0.0f;
}

// ---------------------------------------------------------------------------
// Tensor Data Mover: async-load a `lines` x `line_elems` bf16 2D tile (line
// stride `ld` elements) into LDS at byte offset `lds_off`, appending
// 8 DWORDs (32B) of LDS padding after every line (pad_interval code selects
// the line length in DWORDs: 3 => 16 DW = 64B, 4 => 32 DW = 128B).
// D# layout per CDNA5 ISA.
// ---------------------------------------------------------------------------
static __device__ __forceinline__ void tdm_load_tile(
    unsigned lds_off, const __bf16* gptr, unsigned ld,
    unsigned lines, unsigned line_elems, unsigned pad_int_code) {
    const unsigned long long ga = (unsigned long long)(size_t)gptr;
    u32x4 g0;
    g0.x = 1u;                                   // count=1, user descriptor
    g0.y = lds_off;                              // LDS byte address
    g0.z = (unsigned)(ga & 0xffffffffull);       // global_addr[31:0]
    g0.w = (unsigned)((ga >> 32) & 0x1ffffffull) // global_addr[56:32]
           | (2u << 30);                         // type = image
    i32x8 g1;
    g1[0] = (int)((1u << 16)              // data_size = 2 bytes
                | (1u << 20)              // pad_enable
                | (pad_int_code << 22)    // pad interval (DWORDs per line)
                | (7u << 25));            // pad_amount: 8 DWORDs (32B)
    g1[1] = (int)((ld & 0xffffu) << 16);                    // tensor_dim0 lo
    g1[2] = (int)(((ld >> 16) & 0xffffu) | (lines << 16));  // dim0 hi|dim1 lo
    g1[3] = (int)(line_elems << 16);                        // tile_dim0
    g1[4] = (int)lines;                                     // tile_dim1
    g1[5] = (int)ld;                                        // dim0_stride lo
    g1[6] = 0;
    g1[7] = 0;
    const i32x4 gz4 = {0, 0, 0, 0};              // 2D tensor: groups 2/3 unused
    const i32x8 gz8 = {0, 0, 0, 0, 0, 0, 0, 0};  // extra group (clang-23 form)
    __builtin_amdgcn_tensor_load_to_lds(g0, g1, gz4, gz4, gz8, 0);
}

static __device__ __forceinline__ unsigned lds_offset_of(const void* p) {
    // LDS aperture keeps the workgroup-local offset in the low 32 bits.
    return (unsigned)(size_t)p;
}

// ---------------------------------------------------------------------------
// Embedding gather: x[tok,:] = embed[idx[tok],:]   (fp32 residual master)
// ---------------------------------------------------------------------------
__global__ __launch_bounds__(256) void embed_gather_kernel(
    const int* __restrict__ idx, const float* __restrict__ embed,
    float* __restrict__ x) {
    const int tok = blockIdx.x;
    const int id  = idx[tok];
    const float4* src = (const float4*)(embed + (size_t)id * DMODEL);
    float4*       dst = (float4*)(x + (size_t)tok * DMODEL);
    dst[threadIdx.x] = src[threadIdx.x];
}

// ---------------------------------------------------------------------------
// Straight cast fp32 -> bf16 (for embed, already [N,K] = [V,D])
// ---------------------------------------------------------------------------
__global__ __launch_bounds__(256) void cast_bf16_kernel(
    const float* __restrict__ in, __bf16* __restrict__ out, int n) {
    const int i = blockIdx.x * 256 + threadIdx.x;
    if (i < n) out[i] = (__bf16)in[i];
}

// ---------------------------------------------------------------------------
// Transpose-cast: in fp32 [K,N] row-major -> out bf16 [N,K] row-major.
// ---------------------------------------------------------------------------
__global__ __launch_bounds__(256) void transpose_cast_kernel(
    const float* __restrict__ in, __bf16* __restrict__ out, int K, int N) {
    __shared__ float tile[32][33];
    const int tx = threadIdx.x & 31;
    const int ty = threadIdx.x >> 5;          // 0..7
    const int n0 = blockIdx.x * 32;
    const int k0 = blockIdx.y * 32;
#pragma unroll
    for (int i = 0; i < 32; i += 8)
        tile[ty + i][tx] = in[(size_t)(k0 + ty + i) * N + n0 + tx];
    __syncthreads();
#pragma unroll
    for (int i = 0; i < 32; i += 8)
        out[(size_t)(n0 + ty + i) * K + k0 + tx] = (__bf16)tile[tx][ty + i];
}

// ---------------------------------------------------------------------------
// RMSNorm: y(bf16) = x * rsqrt(mean(x^2)+eps) * w     (one block per row)
// ---------------------------------------------------------------------------
__global__ __launch_bounds__(256) void rmsnorm_kernel(
    const float* __restrict__ x, const float* __restrict__ w,
    __bf16* __restrict__ y) {
    __shared__ float red[256];
    const int row = blockIdx.x;
    const float* xr = x + (size_t)row * DMODEL;
    float ss = 0.0f;
    for (int i = threadIdx.x; i < DMODEL; i += 256) {
        const float v = xr[i];
        ss += v * v;
    }
    red[threadIdx.x] = ss;
    __syncthreads();
    for (int s = 128; s > 0; s >>= 1) {
        if (threadIdx.x < s) red[threadIdx.x] += red[threadIdx.x + s];
        __syncthreads();
    }
    const float scale = rsqrtf(red[0] / (float)DMODEL + EPSRMS);
    __bf16* yr = y + (size_t)row * DMODEL;
    for (int i = threadIdx.x; i < DMODEL; i += 256)
        yr[i] = (__bf16)(xr[i] * scale * w[i]);
}

// ---------------------------------------------------------------------------
// RoPE in place on bf16 [B,T,H,HD]; one thread per rotation pair.
// ---------------------------------------------------------------------------
__global__ __launch_bounds__(256) void rope_kernel(__bf16* __restrict__ x) {
    const int gid = blockIdx.x * blockDim.x + threadIdx.x;
    const int p  = gid & 31;
    const int hh = (gid >> 5) & (NHEAD - 1);
    const int t  = (gid >> 9) & (SEQ - 1);
    const int b  = gid >> 19;
    const size_t base = (((size_t)b * SEQ + t) * NHEAD + hh) * HEADD;
    const float inv = __powf(10000.0f, -(float)(2 * p) / (float)HEADD);
    const float ang = (float)t * inv;
    const float c = __cosf(ang), s = __sinf(ang);
    const float x0 = (float)x[base + p];
    const float x1 = (float)x[base + p + HEADD / 2];
    x[base + p]             = (__bf16)(x0 * c - x1 * s);
    x[base + p + HEADD / 2] = (__bf16)(x1 * c + x0 * s);
}

// ---------------------------------------------------------------------------
// SiLU gate: h1 = silu(h1) * h3   (bf16 in/out, f32 math)
// ---------------------------------------------------------------------------
__global__ __launch_bounds__(256) void silu_gate_kernel(
    __bf16* __restrict__ h1, const __bf16* __restrict__ h3, int n) {
    const int i = blockIdx.x * blockDim.x + threadIdx.x;
    if (i < n) {
        const float a = (float)h1[i];
        h1[i] = (__bf16)((a / (1.0f + __expf(-a))) * (float)h3[i]);
    }
}

// ---------------------------------------------------------------------------
// Tiled WMMA GEMM: C[M,N] = A[M,K](bf16) * B[N,K](bf16, k-major) [+ Res(f32)]
// Block tile 128x128, K-step 32, TDM-staged double-buffered LDS tiles.
// 8 waves arranged 4x2; each wave owns 32x64 = 8 WMMA accumulators.
// ---------------------------------------------------------------------------
#define BM 128
#define BN 128
#define BK 32
#define LDSK (BK + 16)   // 48 bf16 = 96B row stride (32B aligned, TDM pad)

template <bool RES, typename CT>
__global__ __launch_bounds__(256) void gemm_wmma_kernel(
    const __bf16* __restrict__ A, const __bf16* __restrict__ B,
    const float* __restrict__ Res, CT* __restrict__ C,
    int M, int N, int K) {
    __shared__ alignas(32) __bf16 Asb[2][BM][LDSK];
    __shared__ alignas(32) __bf16 Bsb[2][BN][LDSK];

    const int tid  = threadIdx.x;
    const int lane = tid & 31;
    const int wave = tid >> 5;
    const int wm = (wave >> 1) * 32;
    const int wn = (wave & 1) * 64;
    const int lr = lane & 15;
    const int lk = (lane >> 4) * 16;
    const size_t bm = (size_t)blockIdx.y * BM;
    const size_t bn = (size_t)blockIdx.x * BN;

    v8f acc[2][4];
#pragma unroll
    for (int i = 0; i < 2; ++i)
#pragma unroll
        for (int j = 0; j < 4; ++j) zero8(acc[i][j]);

    const __bf16* Ab = A + bm * (size_t)K;
    const __bf16* Bb = B + bn * (size_t)K;
    const unsigned aoff0 = lds_offset_of(&Asb[0][0][0]);
    const unsigned aoff1 = lds_offset_of(&Asb[1][0][0]);
    const unsigned boff0 = lds_offset_of(&Bsb[0][0][0]);
    const unsigned boff1 = lds_offset_of(&Bsb[1][0][0]);

    if (wave == 0) {                         // TDM: preload tile 0
        tdm_load_tile(aoff0, Ab, (unsigned)K, BM, BK, 3);
        tdm_load_tile(boff0, Bb, (unsigned)K, BN, BK, 3);
    }

    const int nk = K / BK;
    int buf = 0;
    for (int t = 0; t < nk; ++t) {
        if (wave == 0) {
            if (t + 1 < nk) {                // TDM: prefetch next tile pair
                const unsigned na = buf ? aoff0 : aoff1;
                const unsigned nb = buf ? boff0 : boff1;
                tdm_load_tile(na, Ab + (size_t)(t + 1) * BK, (unsigned)K,
                              BM, BK, 3);
                tdm_load_tile(nb, Bb + (size_t)(t + 1) * BK, (unsigned)K,
                              BN, BK, 3);
                __builtin_amdgcn_s_wait_tensorcnt(2);  // current pair done
            } else {
                __builtin_amdgcn_s_wait_tensorcnt(0);
            }
        }
        __syncthreads();

        v16bf afrag[2], bfrag[4];
#pragma unroll
        for (int i = 0; i < 2; ++i)
            afrag[i] = *(const v16bf*)&Asb[buf][wm + i * 16 + lr][lk];
#pragma unroll
        for (int j = 0; j < 4; ++j)
            bfrag[j] = *(const v16bf*)&Bsb[buf][wn + j * 16 + lr][lk];
#pragma unroll
        for (int i = 0; i < 2; ++i)
#pragma unroll
            for (int j = 0; j < 4; ++j)
                acc[i][j] = wmma_bf16(afrag[i], bfrag[j], acc[i][j]);

        __syncthreads();                     // buf safe to overwrite next iter
        buf ^= 1;
    }

    const int half = lane >> 4;
    const int ln   = lane & 15;
#pragma unroll
    for (int i = 0; i < 2; ++i)
#pragma unroll
        for (int j = 0; j < 4; ++j)
#pragma unroll
            for (int r = 0; r < 8; ++r) {
                const size_t row = bm + wm + i * 16 + r + half * 8;
                const size_t col = bn + wn + j * 16 + ln;
                float v = acc[i][j][r];
                if (RES) v += Res[row * (size_t)N + col];
                C[row * (size_t)N + col] = (CT)v;
            }
}

// ---------------------------------------------------------------------------
// Flash attention (causal). q/k/v/o are bf16 [B,T,H,HD] (== [B,T,D]).
// Block = 8 waves x 16 query rows = 128 queries per (b,h). The key-tile loop
// is block-uniform: per tile, K(32x64) is TDM-staged and V is transposed
// into LDS cooperatively; waves past their causal bound skip compute but
// keep the barriers. Softmax is wave-private (s_wait_dscnt fences).
// ---------------------------------------------------------------------------
#define ATT_WAVES 8
#define KLDS (HEADD + 16)   // 80 bf16 = 160B row stride (32B aligned)
#define VLDS (32 + 16)      // 48 bf16 = 96B row stride  (32B aligned)

__global__ __launch_bounds__(256) void attention_kernel(
    const __bf16* __restrict__ q, const __bf16* __restrict__ k,
    const __bf16* __restrict__ v, __bf16* __restrict__ o) {
    __shared__ alignas(32) __bf16 Ks[32][KLDS];     // [key][dim]
    __shared__ alignas(32) __bf16 Vs[HEADD][VLDS];  // [dim][key] (transposed)
    __shared__ float              Sb[ATT_WAVES][16][33];
    __shared__ alignas(32) __bf16 Pb[ATT_WAVES][16][32];
    __shared__ float Mw[ATT_WAVES][16];
    __shared__ float Lw[ATT_WAVES][16];
    __shared__ float Aw[ATT_WAVES][16];

    const int tid  = threadIdx.x;
    const int lane = tid & 31;
    const int wave = tid >> 5;
    const int b = blockIdx.z;
    const int h = blockIdx.y;
    const int qblock = blockIdx.x * (ATT_WAVES * 16);
    const int qbase  = qblock + wave * 16;
    const int kmax   = qblock + ATT_WAVES * 16 - 1;  // block-uniform bound
    const int lr = lane & 15;
    const int half = lane >> 4;
    const int lk = half * 16;
    const size_t headoff = (size_t)h * HEADD;

    // Q fragments once (HD=64 -> two K=32 steps), direct 32B bf16 loads.
    v16bf qa[2];
    {
        const __bf16* qp =
            q + ((size_t)(b * SEQ + qbase + lr)) * DMODEL + headoff;
        qa[0] = *(const v16bf*)(qp + lk);
        qa[1] = *(const v16bf*)(qp + 32 + lk);
    }

    if (lane < 16) {
        Mw[wave][lane] = -3.0e38f;
        Lw[wave][lane] = 0.0f;
    }
    wave_lds_fence();

    v8f oacc[4];
#pragma unroll
    for (int nt = 0; nt < 4; ++nt) zero8(oacc[nt]);

    const unsigned ksoff = lds_offset_of(&Ks[0][0]);
    const int vkey  = tid >> 3;          // 0..31: key row this thread stages
    const int vdoff = (tid & 7) * 8;     // 0..56: dim chunk (8 bf16 = 16B)

    for (int kt = 0; kt <= kmax; kt += 32) {
        // ---- cooperative staging: K via TDM, V transposed via LDS ----
        if (wave == 0) {
            tdm_load_tile(ksoff, k + ((size_t)(b * SEQ + kt)) * DMODEL + headoff,
                          DMODEL, 32, HEADD, 4);
        }
        {
            const __bf16* vrow =
                v + ((size_t)(b * SEQ + kt + vkey)) * DMODEL + headoff + vdoff;
            const v8bf vv = *(const v8bf*)vrow;
#pragma unroll
            for (int i = 0; i < 8; ++i) Vs[vdoff + i][vkey] = vv[i];
            if (kt + 32 <= kmax)          // prefetch next V tile rows
                __builtin_prefetch(vrow + 32 * DMODEL, 0, 1);
        }
        if (wave == 0) __builtin_amdgcn_s_wait_tensorcnt(0);
        __syncthreads();

        if (kt <= qbase + 15) {           // wave-uniform causal guard
            // ---- S(16x32) = Q * K^T from LDS, scaled + causal mask ----
#pragma unroll
            for (int nt = 0; nt < 2; ++nt) {
                v8f s;
                zero8(s);
                const __bf16* kr = &Ks[nt * 16 + lr][0];
                s = wmma_bf16(qa[0], *(const v16bf*)(kr + lk), s);
                s = wmma_bf16(qa[1], *(const v16bf*)(kr + 32 + lk), s);
#pragma unroll
                for (int r = 0; r < 8; ++r) {
                    const int m    = r + half * 8;
                    const int kidx = kt + nt * 16 + lr;
                    float val = s[r] * 0.125f;                 // HD^-0.5
                    if (kidx > qbase + m) val = -3.0e38f;      // causal
                    Sb[wave][m][nt * 16 + lr] = val;
                }
            }
            wave_lds_fence();

            // ---- per-row online softmax (lanes 0..15 own one row) ----
            if (lane < 16) {
                const int r = lane;
                float rowmax = -3.0e38f;
                for (int j = 0; j < 32; ++j)
                    rowmax = fmaxf(rowmax, Sb[wave][r][j]);
                const float mold = Mw[wave][r];
                const float mnew = fmaxf(mold, rowmax);
                const float alpha = __expf(mold - mnew);
                float sum = 0.0f;
                for (int j = 0; j < 32; ++j) {
                    const float p = __expf(Sb[wave][r][j] - mnew);
                    Pb[wave][r][j] = (__bf16)p;
                    sum += p;
                }
                Lw[wave][r] = Lw[wave][r] * alpha + sum;
                Mw[wave][r] = mnew;
                Aw[wave][r] = alpha;
            }
            wave_lds_fence();

            // ---- rescale O, then O += P(16x32) * V(32x64) from LDS ----
#pragma unroll
            for (int nt = 0; nt < 4; ++nt)
#pragma unroll
                for (int r = 0; r < 8; ++r)
                    oacc[nt][r] *= Aw[wave][r + half * 8];

            const v16bf pf = *(const v16bf*)&Pb[wave][lr][lk];
#pragma unroll
            for (int nt = 0; nt < 4; ++nt)
                oacc[nt] = wmma_bf16(
                    pf, *(const v16bf*)&Vs[nt * 16 + lr][lk], oacc[nt]);
        }
        __syncthreads();                  // Ks/Vs safe to overwrite
    }

    wave_lds_fence();
#pragma unroll
    for (int nt = 0; nt < 4; ++nt)
#pragma unroll
        for (int r = 0; r < 8; ++r) {
            const int m = r + half * 8;
            o[((size_t)(b * SEQ + qbase + m)) * DMODEL + headoff + nt * 16 +
              lr] = (__bf16)(oacc[nt][r] / Lw[wave][m]);
        }
}

// ---------------------------------------------------------------------------
// Host orchestration. Workspace:
//   x f32 (8MB) | hb bf16 (4MB) | q/k/v/o bf16 (4x4MB) | h1/h3 bf16 (2x16MB)
//   | wbuf bf16 (65.5MB, sized for embed)  => ~126MB total.
// Weight casts and GEMMs are serialized on `stream`, so one wbuf suffices.
// ---------------------------------------------------------------------------
extern "C" void kernel_launch(void* const* d_in, const int* in_sizes, int n_in,
                              void* d_out, int out_size, void* d_ws,
                              size_t ws_size, hipStream_t stream) {
    (void)in_sizes; (void)n_in; (void)out_size; (void)ws_size;
    const int*   idx   = (const int*)d_in[0];
    const float* embed = (const float*)d_in[1];
    const float* Wq    = (const float*)d_in[2];
    const float* Wk    = (const float*)d_in[3];
    const float* Wv    = (const float*)d_in[4];
    const float* Wo    = (const float*)d_in[5];
    const float* W1    = (const float*)d_in[6];
    const float* W2    = (const float*)d_in[7];
    const float* W3    = (const float*)d_in[8];
    const float* n1w   = (const float*)d_in[9];
    const float* n2w   = (const float*)d_in[10];
    const float* fnw   = (const float*)d_in[11];
    float* logits = (float*)d_out;

    const size_t TD = (size_t)MTOK * DMODEL;
    const size_t TF = (size_t)MTOK * FFN;
    char* p = (char*)d_ws;
    float*  x   = (float*)p;   p += TD * 4;
    __bf16* hb  = (__bf16*)p;  p += TD * 2;
    __bf16* qb  = (__bf16*)p;  p += TD * 2;
    __bf16* kb  = (__bf16*)p;  p += TD * 2;
    __bf16* vb  = (__bf16*)p;  p += TD * 2;
    __bf16* ob  = (__bf16*)p;  p += TD * 2;
    __bf16* h1  = (__bf16*)p;  p += TF * 2;
    __bf16* h3  = (__bf16*)p;  p += TF * 2;
    __bf16* wbuf = (__bf16*)p; // VOCAB*DMODEL bf16 (max weight)

    embed_gather_kernel<<<MTOK, 256, 0, stream>>>(idx, embed, x);

    const dim3 gD(DMODEL / BN, MTOK / BM);   // 8 x 16
    const dim3 gF(FFN / BN, MTOK / BM);      // 32 x 16
    const dim3 tDD(DMODEL / 32, DMODEL / 32);
    const dim3 tDF(FFN / 32, DMODEL / 32);   // in [D,F]
    const dim3 tFD(DMODEL / 32, FFN / 32);   // in [F,D]
    const int ropeN = BATCH * SEQ * NHEAD * (HEADD / 2);

    for (int l = 0; l < LAYERS; ++l) {
        const float* wq = Wq + (size_t)l * DMODEL * DMODEL;
        const float* wk = Wk + (size_t)l * DMODEL * DMODEL;
        const float* wv = Wv + (size_t)l * DMODEL * DMODEL;
        const float* wo = Wo + (size_t)l * DMODEL * DMODEL;
        const float* w1 = W1 + (size_t)l * DMODEL * FFN;
        const float* w2 = W2 + (size_t)l * FFN * DMODEL;
        const float* w3 = W3 + (size_t)l * DMODEL * FFN;

        rmsnorm_kernel<<<MTOK, 256, 0, stream>>>(x, n1w + (size_t)l * DMODEL, hb);

        transpose_cast_kernel<<<tDD, 256, 0, stream>>>(wq, wbuf, DMODEL, DMODEL);
        gemm_wmma_kernel<false, __bf16><<<gD, 256, 0, stream>>>(
            hb, wbuf, nullptr, qb, MTOK, DMODEL, DMODEL);
        transpose_cast_kernel<<<tDD, 256, 0, stream>>>(wk, wbuf, DMODEL, DMODEL);
        gemm_wmma_kernel<false, __bf16><<<gD, 256, 0, stream>>>(
            hb, wbuf, nullptr, kb, MTOK, DMODEL, DMODEL);
        transpose_cast_kernel<<<tDD, 256, 0, stream>>>(wv, wbuf, DMODEL, DMODEL);
        gemm_wmma_kernel<false, __bf16><<<gD, 256, 0, stream>>>(
            hb, wbuf, nullptr, vb, MTOK, DMODEL, DMODEL);

        rope_kernel<<<ropeN / 256, 256, 0, stream>>>(qb);
        rope_kernel<<<ropeN / 256, 256, 0, stream>>>(kb);

        attention_kernel<<<dim3(SEQ / (ATT_WAVES * 16), NHEAD, BATCH), 256, 0,
                           stream>>>(qb, kb, vb, ob);

        transpose_cast_kernel<<<tDD, 256, 0, stream>>>(wo, wbuf, DMODEL, DMODEL);
        gemm_wmma_kernel<true, float><<<gD, 256, 0, stream>>>(
            ob, wbuf, x, x, MTOK, DMODEL, DMODEL);     // x += o @ Wo

        rmsnorm_kernel<<<MTOK, 256, 0, stream>>>(x, n2w + (size_t)l * DMODEL, hb);

        transpose_cast_kernel<<<tDF, 256, 0, stream>>>(w1, wbuf, DMODEL, FFN);
        gemm_wmma_kernel<false, __bf16><<<gF, 256, 0, stream>>>(
            hb, wbuf, nullptr, h1, MTOK, FFN, DMODEL);
        transpose_cast_kernel<<<tDF, 256, 0, stream>>>(w3, wbuf, DMODEL, FFN);
        gemm_wmma_kernel<false, __bf16><<<gF, 256, 0, stream>>>(
            hb, wbuf, nullptr, h3, MTOK, FFN, DMODEL);

        silu_gate_kernel<<<(int)((TF + 255) / 256), 256, 0, stream>>>(
            h1, h3, (int)TF);

        transpose_cast_kernel<<<tFD, 256, 0, stream>>>(w2, wbuf, FFN, DMODEL);
        gemm_wmma_kernel<true, float><<<gD, 256, 0, stream>>>(
            h1, wbuf, x, x, MTOK, DMODEL, FFN);        // x += g @ W2
    }

    rmsnorm_kernel<<<MTOK, 256, 0, stream>>>(x, fnw, hb);

    // logits = h @ embed^T : embed is already [N,K] = [V,D]; straight cast.
    const int embN = VOCAB * DMODEL;
    cast_bf16_kernel<<<(embN + 255) / 256, 256, 0, stream>>>(embed, wbuf, embN);
    const dim3 gV(VOCAB / BN, MTOK / BM);    // 250 x 16
    gemm_wmma_kernel<false, float><<<gV, 256, 0, stream>>>(
        hb, wbuf, nullptr, logits, MTOK, VOCAB, DMODEL);
}
